// GCNBlock_44057774522589
// MI455X (gfx1250) — matrix operand order repaired and checked by reading, not compiled, not run on hardware
//
#include <hip/hip_runtime.h>

typedef __attribute__((ext_vector_type(16))) __bf16 v16bf;
typedef __attribute__((ext_vector_type(8)))  float  v8f;
typedef __attribute__((ext_vector_type(4)))  float  f4;   // clang ext vector (nontemporal-load OK)

#define N_NODES 16384
#define D 128
#define EPS 1e-5f
#define SLOPE 0.2f

// fp32 pair -> packed bf16 dword (round-half-up, cheap: 2 adds + 1 perm)
__device__ __forceinline__ unsigned f2bf_pk(float f0, float f1) {
    unsigned u0 = __float_as_uint(f0) + 0x8000u;
    unsigned u1 = __float_as_uint(f1) + 0x8000u;
#if __has_builtin(__builtin_amdgcn_perm)
    // D = {u1.b3, u1.b2, u0.b3, u0.b2}; S0=u1 (bytes 4-7), S1=u0 (bytes 0-3)
    return __builtin_amdgcn_perm(u1, u0, 0x07060302u);
#else
    return (u0 >> 16) | (u1 & 0xFFFF0000u);
#endif
}

__device__ __forceinline__ unsigned short f2bf1(float f) {
    return (unsigned short)((__float_as_uint(f) + 0x8000u) >> 16);
}

// Kernel 1: XW = x @ W  (fp32 accumulate), store transposed bf16: XWT[j][i]
__global__ void __launch_bounds__(D) xw_kernel(const float* __restrict__ x,
                                               const float* __restrict__ W,
                                               unsigned short* __restrict__ XWT) {
    __shared__ float xs[D];
    const int i = blockIdx.x;
    const int j = threadIdx.x;
    xs[j] = x[(size_t)i * D + j];
    __syncthreads();
    float s = 0.f;
#pragma unroll 8
    for (int k = 0; k < D; ++k) s += xs[k] * W[k * D + j];
    XWT[(size_t)j * N_NODES + i] = f2bf1(s);
}

// ---------------- Kernel 2: H = A @ XW + b ----------------
// Block = 4 waves (128 thr): wave w -> row strip (w>>1), column half (w&1).
// Each wave: 16 rows x 64 cols = 4 accumulator tiles, K pipelined in steps
// of 32 with explicit ping/pong register buffers (x2-unrolled loop).

struct Abuf { f4 a0, a1, a2, a3; };
struct Bbuf { uint4 q[8]; };

__device__ __forceinline__ void loadA(Abuf& r, const float* __restrict__ Arow,
                                      int k0, int half) {
    const f4* p0 = (const f4*)(Arow + k0 + half * 8);
    const f4* p1 = (const f4*)(Arow + k0 + 16 + half * 8);
    r.a0 = __builtin_nontemporal_load(p0);
    r.a1 = __builtin_nontemporal_load(p0 + 1);
    r.a2 = __builtin_nontemporal_load(p1);
    r.a3 = __builtin_nontemporal_load(p1 + 1);
}

__device__ __forceinline__ void loadB(Bbuf& r, const unsigned short* __restrict__ Bbase,
                                      int k0) {
#pragma unroll
    for (int t = 0; t < 4; ++t) {
        const uint4* pb = (const uint4*)(Bbase + (size_t)t * 16 * N_NODES + k0);
        r.q[2 * t]     = pb[0];
        r.q[2 * t + 1] = pb[2];
    }
}

__device__ __forceinline__ void compute(const Abuf& ab, const Bbuf& bb, v8f c[4]) {
    // A fragment (16x32 bf16), ISA layout:
    // lanes 0-15: K = +0..7 (elems 0..7), +16..23 (elems 8..15)
    // lanes 16-31: K = +8..15, +24..31
    union { unsigned u[8]; v16bf v; } afrag;
    afrag.u[0] = f2bf_pk(ab.a0.x, ab.a0.y);
    afrag.u[1] = f2bf_pk(ab.a0.z, ab.a0.w);
    afrag.u[2] = f2bf_pk(ab.a1.x, ab.a1.y);
    afrag.u[3] = f2bf_pk(ab.a1.z, ab.a1.w);
    afrag.u[4] = f2bf_pk(ab.a2.x, ab.a2.y);
    afrag.u[5] = f2bf_pk(ab.a2.z, ab.a2.w);
    afrag.u[6] = f2bf_pk(ab.a3.x, ab.a3.y);
    afrag.u[7] = f2bf_pk(ab.a3.z, ab.a3.w);
#pragma unroll
    for (int t = 0; t < 4; ++t) {
        union { uint4 q[2]; v16bf v; } bfrag;
        bfrag.q[0] = bb.q[2 * t];
        bfrag.q[1] = bb.q[2 * t + 1];
        c[t] = __builtin_amdgcn_wmma_f32_16x16x32_bf16(
            false, afrag.v, false, bfrag.v, (short)0, c[t], false, false);
    }
}

__global__ void __launch_bounds__(128) gemm_kernel(const float* __restrict__ A,
                                                   const unsigned short* __restrict__ XWT,
                                                   const float* __restrict__ bias,
                                                   float* __restrict__ H) {
    const int wave = threadIdx.x >> 5;
    const int lane = threadIdx.x & 31;
    const int half = lane >> 4;
    const int l    = lane & 15;
    const int strip = wave >> 1;       // 0..1
    const int h2    = wave & 1;        // column half: 0..1
    const int m0    = blockIdx.x * 32 + strip * 16;
    const int nbase = h2 * 64;

    const float* Arow = A + (size_t)(m0 + l) * N_NODES;
    const unsigned short* Bbase = XWT + (size_t)(nbase + l) * N_NODES + half * 8;

    v8f c[4];
#pragma unroll
    for (int t = 0; t < 4; ++t)
#pragma unroll
        for (int e = 0; e < 8; ++e) c[t][e] = 0.f;

    Abuf aP, aQ;
    Bbuf bP, bQ;
    loadA(aP, Arow, 0, half);
    loadB(bP, Bbase, 0);

    int k0 = 0;
    for (; k0 < N_NODES - 64; k0 += 64) {
        loadA(aQ, Arow, k0 + 32, half);
        loadB(bQ, Bbase, k0 + 32);
        compute(aP, bP, c);            // consumes loads issued one stage ago
        loadA(aP, Arow, k0 + 64, half);
        loadB(bP, Bbase, k0 + 64);
        compute(aQ, bQ, c);
    }
    // k0 == N_NODES - 64: last two stages, prefetch only the final one
    loadA(aQ, Arow, k0 + 32, half);
    loadB(bQ, Bbase, k0 + 32);
    compute(aP, bP, c);
    compute(aQ, bQ, c);

    // store C (+bias): lane l -> N = nbase + t*16 + l; VGPR r -> M = r + half*8
#pragma unroll
    for (int t = 0; t < 4; ++t) {
        const int n = nbase + t * 16 + l;
        const float bv = bias[n];
#pragma unroll
        for (int r = 0; r < 8; ++r) {
            const int m = m0 + half * 8 + r;
            H[(size_t)m * D + n] = c[t][r] + bv;
        }
    }
}

// Kernel 3: per-column mean and rsqrt(var + eps)  (biased variance)
__global__ void __launch_bounds__(256) stats_kernel(const float* __restrict__ H,
                                                    float* __restrict__ mean,
                                                    float* __restrict__ rstd) {
    __shared__ float ssum[256];
    __shared__ float ssq[256];
    const int j = blockIdx.x;
    float s = 0.f, q = 0.f;
    for (int i = threadIdx.x; i < N_NODES; i += 256) {
        float h = H[(size_t)i * D + j];
        s += h;
        q += h * h;
    }
    ssum[threadIdx.x] = s;
    ssq[threadIdx.x]  = q;
    __syncthreads();
    for (int off = 128; off > 0; off >>= 1) {
        if (threadIdx.x < off) {
            ssum[threadIdx.x] += ssum[threadIdx.x + off];
            ssq[threadIdx.x]  += ssq[threadIdx.x + off];
        }
        __syncthreads();
    }
    if (threadIdx.x == 0) {
        float m   = ssum[0] * (1.0f / N_NODES);
        float var = ssq[0] * (1.0f / N_NODES) - m * m;
        mean[j] = m;
        rstd[j] = rsqrtf(var + EPS);
    }
}

// Kernel 4: y = gamma*(h-mean)*rstd + beta ; LeakyReLU(0.2)
__global__ void __launch_bounds__(256) bn_kernel(const float* __restrict__ H,
                                                 const float* __restrict__ gamma,
                                                 const float* __restrict__ beta,
                                                 const float* __restrict__ mean,
                                                 const float* __restrict__ rstd,
                                                 float* __restrict__ out) {
    const int idx = blockIdx.x * 256 + threadIdx.x;
    const int j = idx & (D - 1);
    float h = H[idx];
    float y = gamma[j] * (h - mean[j]) * rstd[j] + beta[j];
    out[idx] = (y >= 0.f) ? y : SLOPE * y;
}

extern "C" void kernel_launch(void* const* d_in, const int* in_sizes, int n_in,
                              void* d_out, int out_size, void* d_ws, size_t ws_size,
                              hipStream_t stream) {
    const float* x     = (const float*)d_in[0];
    const float* A     = (const float*)d_in[1];
    const float* W     = (const float*)d_in[2];
    const float* b     = (const float*)d_in[3];
    const float* gamma = (const float*)d_in[4];
    const float* beta  = (const float*)d_in[5];
    float* out = (float*)d_out;

    char* ws = (char*)d_ws;
    unsigned short* XWT = (unsigned short*)ws;                                  // 4 MB bf16 [D][N]
    float* H    = (float*)(ws + (size_t)D * N_NODES * sizeof(unsigned short));  // 8 MB
    float* mean = (float*)(ws + (size_t)D * N_NODES * sizeof(unsigned short)
                              + (size_t)N_NODES * D * sizeof(float));
    float* rstd = mean + D;

    xw_kernel<<<N_NODES, D, 0, stream>>>(x, W, XWT);
    gemm_kernel<<<N_NODES / 32, 128, 0, stream>>>(A, XWT, b, H);
    stats_kernel<<<D, 256, 0, stream>>>(H, mean, rstd);
    bn_kernel<<<(N_NODES * D) / 256, 256, 0, stream>>>(H, gamma, beta, mean, rstd, out);
}
